// LongAttentionCondenser_51058571215286
// MI455X (gfx1250) — matrix operand order repaired and checked by reading, not compiled, not run on hardware
//
#include <hip/hip_runtime.h>
#include <hip/hip_bf16.h>
#include <hip/hip_fp16.h>

typedef __attribute__((ext_vector_type(16))) __bf16 v16bf;
typedef __attribute__((ext_vector_type(8)))  float  v8f;
typedef __attribute__((ext_vector_type(4)))  unsigned int v4u;

#define DMODEL   1024
#define SEQLEN   65536
#define NHEAD    8
#define HEADD    128
#define MTOK     8
#define CHCOL    64          // (h,m) columns = NHEAD*MTOK
#define SCHUNK   32          // sequence rows per chunk
#define HPADW    1028        // padded halves per hbuf row (bank-conflict pad, 8B aligned)
#define HIDD     2048

// ---- workspace layout (bytes) ----
#define OFF_QS       0u           // 8*1024 f32
#define OFF_QWPACK   32768u       // 128 fragments * 1024B (bf16, WMMA-B fragment order)
#define OFF_QB       163840u      // 64 f32 (pad to 4K)
#define OFF_LG       167936u      // 64 f32 (pad)
#define OFF_EG       172032u      // 64*1024 f32
#define OFF_OUTPRE   434176u      // 8*1024 f32
#define OFF_HN       466944u      // 8*1024 f32
#define OFF_H1       499712u      // 8*2048 f32
#define OFF_MPART    565248u      // <=256*64 f32
#define OFF_LPART    630784u      // <=256*64 f32
#define OFF_EPART    696320u      // nblk * 256KiB  (E fragment partials)
#define EPART_FLOATS_PER_BLK 65536u
#define EPART_BYTES_PER_BLK  262144u

static __device__ __forceinline__ unsigned short f2bf(float f) {
    union { float f; unsigned u; } a; a.f = f;
    unsigned r = a.u + 0x7fffu + ((a.u >> 16) & 1u);   // RNE
    return (unsigned short)(r >> 16);
}

// ---------------- K0a: q = (latents @ wq + bq) * HD^-0.5 ----------------
__global__ void k_qproj(const float* __restrict__ lat, const float* __restrict__ wq,
                        const float* __restrict__ bq, float* __restrict__ qs) {
    int gid = blockIdx.x * 256 + threadIdx.x;           // 8192
    int m = gid >> 10, d = gid & 1023;
    float acc = bq[d];
    for (int c = 0; c < DMODEL; ++c) acc += lat[m * DMODEL + c] * wq[(size_t)c * DMODEL + d];
    qs[gid] = acc * 0.08838834764831845f;               // 128^-0.5
}

// ---- K0b: qw[hm,c] = sum_d q[m,h*128+d]*wk[c,h*128+d]; pack to WMMA-B frags; qb ----
__global__ void k_qwpack(const float* __restrict__ qs, const float* __restrict__ wk,
                         const float* __restrict__ bk, unsigned short* __restrict__ qwpack,
                         float* __restrict__ qb) {
    int hm = blockIdx.x;                                 // 0..63, hm = h*8+m
    int h = hm >> 3, m = hm & 7;
    __shared__ float qrow[HEADD];
    int t = threadIdx.x;
    if (t < HEADD) qrow[t] = qs[m * DMODEL + h * HEADD + t];
    __syncthreads();
    for (int c = t; c < DMODEL; c += 256) {
        const float* wr = wk + (size_t)c * DMODEL + h * HEADD;
        float acc = 0.f;
        for (int d2 = 0; d2 < HEADD; ++d2) acc += qrow[d2] * wr[d2];
        int kk = c >> 5, kl = c & 31;
        int lane = ((kl >> 3) & 1) * 16 + (hm & 15);
        int j = (kl & 7) + ((kl >> 4) << 3);
        qwpack[(((kk * 4 + (hm >> 4)) * 32 + lane) << 4) + j] = f2bf(acc);
    }
    if (t == 0) {
        float acc = 0.f;
        for (int d2 = 0; d2 < HEADD; ++d2) acc += qrow[d2] * bk[h * HEADD + d2];
        qb[hm] = acc;
    }
}

// ---------------- K1: fused streaming pass (scores + online softmax + E accum) ----------------
// 512 threads = 16 waves. Each wave owns 16 E tiles in VGPRs (128 regs) -> no spills.
__global__ __launch_bounds__(512, 1)
void k_main(const float* __restrict__ hseq, const unsigned short* __restrict__ qwpack,
            const float* __restrict__ qb, float* __restrict__ mpart, float* __restrict__ lpart,
            float* __restrict__ epart, int chunks) {
    __shared__ __attribute__((aligned(16))) unsigned short hbuf[SCHUNK * HPADW];   // 64.25 KB bf16
    __shared__ float sred[32 * 256];                                               // 32 KB (4ct x 4kq x 2rt)
    __shared__ __attribute__((aligned(16))) unsigned short eafrag[4 * 32 * 16];    // 4 KB
    __shared__ float alphas_s[CHCOL];
    __shared__ float qb_s[CHCOL];
    __shared__ int   rescale_flag;

    const int tid = threadIdx.x, wave = tid >> 5, lane = tid & 31;
    const int blk = blockIdx.x;
    if (tid < CHCOL) qb_s[tid] = qb[tid];
    const int ct = wave & 3, kq = wave >> 2;     // scores: 4 col-tiles x 4-way K split

    v8f E[4][4];                                  // [cl][rt], indices always constant
#pragma unroll
    for (int a = 0; a < 4; ++a)
#pragma unroll
        for (int b = 0; b < 4; ++b)
#pragma unroll
            for (int j = 0; j < 8; ++j) E[a][b][j] = 0.f;

    float m_run = -3.0e38f, l_run = 0.f;
    const float* hbase = hseq + (size_t)blk * chunks * SCHUNK * DMODEL;
    const unsigned long long* hb64 = (const unsigned long long*)hbuf;
    const int r15 = lane & 15, hi8 = (lane >> 4) * 8;

    for (int c0 = 0; c0 < chunks; ++c0) {
        const float4* src = (const float4*)(hbase + (size_t)c0 * SCHUNK * DMODEL);
        if (c0 + 1 < chunks) {
            const char* nxt = (const char*)src + SCHUNK * DMODEL * 4;
            __builtin_prefetch(nxt + tid * 256, 0, 0);
            __builtin_prefetch(nxt + tid * 256 + 128, 0, 0);
        }
        if (tid == 0) rescale_flag = 0;
        // ---- load chunk fp32 -> bf16 -> LDS (row-major, padded) ----
#pragma unroll 8
        for (int i = 0; i < 16; ++i) {
            int idx = tid + i * 512;                     // 8192 float4 = 32x1024 f32
            float4 v = src[idx];
            int e4 = idx << 2, row = e4 >> 10, col = e4 & 1023;
            unsigned long long p = (unsigned long long)f2bf(v.x)
                                 | ((unsigned long long)f2bf(v.y) << 16)
                                 | ((unsigned long long)f2bf(v.z) << 32)
                                 | ((unsigned long long)f2bf(v.w) << 48);
            *(unsigned long long*)&hbuf[row * HPADW + col] = p;
        }
        __syncthreads();

        // ---- scores: [32 seq x 64 hm] = h_chunk @ qwT ; K=1024 split 4 ways ----
        v8f c0a, c1a;
#pragma unroll
        for (int j = 0; j < 8; ++j) { c0a[j] = 0.f; c1a[j] = 0.f; }
#pragma unroll 2
        for (int kt = 0; kt < 8; ++kt) {
            int kk = kq * 8 + kt;
            union { v16bf v; v4u u2[2]; } bq_;
            const v4u* bp = (const v4u*)(qwpack + (((kk * 4 + ct) * 32 + lane) << 4));
            bq_.u2[0] = bp[0]; bq_.u2[1] = bp[1];
            union { v16bf v; unsigned long long q[4]; } a0, a1;
            int kb = kk * 32 + hi8;
            unsigned ia = (unsigned)(r15 * HPADW + kb) >> 2;
            a0.q[0] = hb64[ia];     a0.q[1] = hb64[ia + 1];
            a0.q[2] = hb64[ia + 4]; a0.q[3] = hb64[ia + 5];
            unsigned ib = (unsigned)((16 + r15) * HPADW + kb) >> 2;
            a1.q[0] = hb64[ib];     a1.q[1] = hb64[ib + 1];
            a1.q[2] = hb64[ib + 4]; a1.q[3] = hb64[ib + 5];
            c0a = __builtin_amdgcn_wmma_f32_16x16x32_bf16(false, a0.v, false, bq_.v, (short)0, c0a, false, false);
            c1a = __builtin_amdgcn_wmma_f32_16x16x32_bf16(false, a1.v, false, bq_.v, (short)0, c1a, false, false);
        }
        {   // stash K-partials (j-major => conflict-free b32)
            int slot0 = ((ct * 4 + kq) * 2) * 256;
#pragma unroll
            for (int j = 0; j < 8; ++j) {
                sred[slot0 + j * 32 + lane] = c0a[j];
                sred[slot0 + 256 + j * 32 + lane] = c1a[j];
            }
        }
        __syncthreads();

        // ---- waves 0-3: reduce K-split, online softmax, emit e^T A-fragments ----
        if (wave < 4) {
            v8f s0, s1;
#pragma unroll
            for (int j = 0; j < 8; ++j) { s0[j] = 0.f; s1[j] = 0.f; }
#pragma unroll
            for (int q2 = 0; q2 < 4; ++q2) {
                int sl = ((wave * 4 + q2) * 2) * 256;
#pragma unroll
                for (int j = 0; j < 8; ++j) {
                    s0[j] += sred[sl + j * 32 + lane];
                    s1[j] += sred[sl + 256 + j * 32 + lane];
                }
            }
            float qbv = qb_s[wave * 16 + r15];
            float cmax = -3.0e38f;
#pragma unroll
            for (int j = 0; j < 8; ++j) {
                s0[j] += qbv; s1[j] += qbv;
                cmax = fmaxf(cmax, fmaxf(s0[j], s1[j]));
            }
            cmax = fmaxf(cmax, __shfl_xor(cmax, 16));
            float nm = fmaxf(m_run, cmax);
            float alpha = __expf(m_run - nm);            // == 1.0 exactly when max unchanged
            float lsum = 0.f;
            unsigned short eh[16];
#pragma unroll
            for (int j = 0; j < 8; ++j) {
                float e0 = __expf(s0[j] - nm), e1 = __expf(s1[j] - nm);
                lsum += e0 + e1;
                eh[j] = f2bf(e0); eh[8 + j] = f2bf(e1);
            }
            lsum += __shfl_xor(lsum, 16);
            l_run = l_run * alpha + lsum;
            m_run = nm;
            if (lane < 16) {
                alphas_s[wave * 16 + lane] = alpha;
                if (alpha < 1.0f) atomicOr(&rescale_flag, 1);
            }
            unsigned long long* ea = (unsigned long long*)&eafrag[(wave * 32 + lane) << 4];
#pragma unroll
            for (int k2 = 0; k2 < 4; ++k2) {
                unsigned long long p = (unsigned long long)eh[k2 * 4]
                                     | ((unsigned long long)eh[k2 * 4 + 1] << 16)
                                     | ((unsigned long long)eh[k2 * 4 + 2] << 32)
                                     | ((unsigned long long)eh[k2 * 4 + 3] << 48);
                ea[k2] = p;
            }
        }
        __syncthreads();

        // ---- in-place alpha rescale of E (block-uniform skip when no max changed) ----
        if (rescale_flag) {
            int abase0 = (lane < 16) ? 0 : 8;
#pragma unroll
            for (int rt = 0; rt < 4; ++rt) {
                float al[8];
#pragma unroll
                for (int j = 0; j < 8; ++j) al[j] = alphas_s[rt * 16 + abase0 + j];
#pragma unroll
                for (int cl = 0; cl < 4; ++cl)
#pragma unroll
                    for (int j = 0; j < 8; ++j) E[cl][rt][j] *= al[j];
            }
        }

        // ---- E += e^T @ h_chunk  (E stays in VGPRs) ----
        const unsigned long long* eaf = (const unsigned long long*)eafrag;
        int k0 = (lane >> 4) * 8;
#pragma unroll
        for (int cl = 0; cl < 4; ++cl) {
            int ctc = wave * 4 + cl;
            int coln = ctc * 16 + r15;
            unsigned tmpu[16];
#pragma unroll
            for (int j = 0; j < 16; ++j) tmpu[j] = hbuf[(k0 + j + (j & 8)) * HPADW + coln];  // 16 independent ds_load_u16
            union { v16bf v; unsigned u32[8]; } bb;
#pragma unroll
            for (int j = 0; j < 8; ++j) bb.u32[j] = tmpu[2 * j] | (tmpu[2 * j + 1] << 16);
#pragma unroll
            for (int rt = 0; rt < 4; ++rt) {
                union { v16bf v; unsigned long long q[4]; } aa;
                int ib3 = (rt * 32 + lane) * 4;
                aa.q[0] = eaf[ib3]; aa.q[1] = eaf[ib3 + 1];
                aa.q[2] = eaf[ib3 + 2]; aa.q[3] = eaf[ib3 + 3];
                E[cl][rt] = __builtin_amdgcn_wmma_f32_16x16x32_bf16(false, aa.v, false, bb.v, (short)0, E[cl][rt], false, false);
            }
        }
        __syncthreads();
    }

    // ---- write per-block partials ----
    if (wave < 4 && lane < 16) {
        int idx = blk * CHCOL + wave * 16 + lane;
        mpart[idx] = m_run; lpart[idx] = l_run;
    }
    float* eb = epart + (size_t)blk * EPART_FLOATS_PER_BLK;
#pragma unroll
    for (int cl = 0; cl < 4; ++cl) {
        int ctc = wave * 4 + cl;
#pragma unroll
        for (int rt = 0; rt < 4; ++rt) {
            int t2 = rt * 64 + ctc;
            union { v8f v; v4u u[2]; } w2; w2.v = E[cl][rt];
            v4u* dst = (v4u*)(eb + (size_t)(t2 * 32 + lane) * 8);
            dst[0] = w2.u[0]; dst[1] = w2.u[1];
        }
    }
}

// ---------------- K2: log-sum-exp merge of block partials ----------------
__global__ void k_merge(const float* __restrict__ mpart, const float* __restrict__ lpart,
                        const float* __restrict__ epart, float* __restrict__ eg,
                        float* __restrict__ lg, int nblk) {
    int hm = blockIdx.x, t = threadIdx.x;
    __shared__ float red[256];
    __shared__ float scl[256];
    float mv = (t < nblk) ? mpart[t * CHCOL + hm] : -3.0e38f;
    red[t] = mv; __syncthreads();
    for (int s = 128; s > 0; s >>= 1) { if (t < s) red[t] = fmaxf(red[t], red[t + s]); __syncthreads(); }
    float mg = red[0]; __syncthreads();
    float sv = (t < nblk) ? __expf(mpart[t * CHCOL + hm] - mg) : 0.f;
    scl[t] = sv;
    red[t] = (t < nblk) ? sv * lpart[t * CHCOL + hm] : 0.f;
    __syncthreads();
    for (int s = 128; s > 0; s >>= 1) { if (t < s) red[t] += red[t + s]; __syncthreads(); }
    if (t == 0) lg[hm] = red[0];
    int rt = hm >> 4, rr = hm & 15, j = rr & 7;
    int laneoff = (rr >= 8) ? 16 : 0;
    for (int cc = 0; cc < 4; ++cc) {
        int c = t + cc * 256;
        int t2 = rt * 64 + (c >> 4);
        size_t off = (size_t)(t2 * 32 + (c & 15) + laneoff) * 8 + j;
        float acc = 0.f;
        for (int b = 0; b < nblk; ++b) acc += scl[b] * epart[(size_t)b * EPART_FLOATS_PER_BLK + off];
        eg[hm * DMODEL + c] = acc;
    }
}

// ---------------- epilogue kernels ----------------
__global__ void k_attnout(const float* __restrict__ eg, const float* __restrict__ lg,
                          const float* __restrict__ wv, const float* __restrict__ bv,
                          float* __restrict__ outpre) {
    int gid = blockIdx.x * 256 + threadIdx.x;           // 8192
    int m = gid >> 10, dcol = gid & 1023;
    int hm = (dcol >> 7) * 8 + m;
    const float* er = eg + hm * DMODEL;
    float acc = 0.f;
    for (int c = 0; c < DMODEL; ++c) acc += er[c] * wv[(size_t)c * DMODEL + dcol];
    float l = lg[hm];
    outpre[gid] = (acc + l * bv[dcol]) / (l + 1e-9f);
}

__global__ void k_oproj(const float* __restrict__ outpre, const float* __restrict__ wo,
                        const float* __restrict__ bo, float* __restrict__ out1) {
    int gid = blockIdx.x * 256 + threadIdx.x;           // 8192
    int m = gid >> 10, d = gid & 1023;
    float acc = bo[d];
    for (int c = 0; c < DMODEL; ++c) acc += outpre[m * DMODEL + c] * wo[(size_t)c * DMODEL + d];
    out1[gid] = acc;
}

__global__ void k_ln(const float* __restrict__ out1, const float* __restrict__ g,
                     const float* __restrict__ b, float* __restrict__ hn) {
    int m = blockIdx.x, t = threadIdx.x;
    __shared__ float s1[256], s2[256];
    float a = 0.f, a2 = 0.f;
    for (int c = t; c < DMODEL; c += 256) { float x = out1[m * DMODEL + c]; a += x; a2 += x * x; }
    s1[t] = a; s2[t] = a2; __syncthreads();
    for (int s = 128; s > 0; s >>= 1) { if (t < s) { s1[t] += s1[t + s]; s2[t] += s2[t + s]; } __syncthreads(); }
    float mu = s1[0] * (1.f / DMODEL);
    float var = s2[0] * (1.f / DMODEL) - mu * mu;
    float rs = rsqrtf(var + 1e-5f);
    for (int c = t; c < DMODEL; c += 256) {
        float x = out1[m * DMODEL + c];
        hn[m * DMODEL + c] = (x - mu) * rs * g[c] + b[c];
    }
}

__global__ void k_mlp1(const float* __restrict__ hn, const float* __restrict__ w1,
                       const float* __restrict__ b1, float* __restrict__ h1) {
    int gid = blockIdx.x * 256 + threadIdx.x;           // 16384
    int m = gid >> 11, j = gid & 2047;
    float acc = b1[j];
    for (int c = 0; c < DMODEL; ++c) acc += hn[m * DMODEL + c] * w1[(size_t)c * HIDD + j];
    h1[gid] = 0.5f * acc * (1.f + erff(acc * 0.70710678118654752f));
}

__global__ void k_mlp2(const float* __restrict__ h1, const float* __restrict__ w2,
                       const float* __restrict__ b2, float* __restrict__ dout) {
    int gid = blockIdx.x * 256 + threadIdx.x;           // 8192
    int m = gid >> 10, d = gid & 1023;
    float acc = b2[d];
    for (int j = 0; j < HIDD; ++j) acc += h1[m * HIDD + j] * w2[(size_t)j * DMODEL + d];
    dout[gid] += acc;                                    // residual over out1 already in d_out
}

__global__ void k_mvec(const float* __restrict__ dout, float* __restrict__ mv) {
    int d = blockIdx.x * 256 + threadIdx.x;              // 1024
    float a = 0.f;
    for (int m = 0; m < MTOK; ++m) a += dout[m * DMODEL + d];
    mv[d] = a * 0.125f;
}

extern "C" void kernel_launch(void* const* d_in, const int* in_sizes, int n_in,
                              void* d_out, int out_size, void* d_ws, size_t ws_size,
                              hipStream_t stream) {
    const float* hseq = (const float*)d_in[0];
    const float* lat  = (const float*)d_in[1];
    const float* wq   = (const float*)d_in[2];  const float* bq = (const float*)d_in[3];
    const float* wk   = (const float*)d_in[4];  const float* bk = (const float*)d_in[5];
    const float* wv   = (const float*)d_in[6];  const float* bv = (const float*)d_in[7];
    const float* wo   = (const float*)d_in[8];  const float* bo = (const float*)d_in[9];
    const float* lng  = (const float*)d_in[10]; const float* lnb = (const float*)d_in[11];
    const float* w1   = (const float*)d_in[12]; const float* b1 = (const float*)d_in[13];
    const float* w2   = (const float*)d_in[14]; const float* b2 = (const float*)d_in[15];
    char* ws = (char*)d_ws;
    float* dout = (float*)d_out;

    // size the streaming grid from available workspace (power of two, <=256)
    int nblk = 1;
    while (nblk < 256 &&
           (size_t)OFF_EPART + (size_t)(nblk * 2) * EPART_BYTES_PER_BLK <= ws_size)
        nblk <<= 1;
    int chunks = (SEQLEN / nblk) / SCHUNK;

    float*          qs     = (float*)(ws + OFF_QS);
    unsigned short* qwpack = (unsigned short*)(ws + OFF_QWPACK);
    float*          qbp    = (float*)(ws + OFF_QB);
    float*          lg     = (float*)(ws + OFF_LG);
    float*          eg     = (float*)(ws + OFF_EG);
    float*          outpre = (float*)(ws + OFF_OUTPRE);
    float*          hn     = (float*)(ws + OFF_HN);
    float*          h1     = (float*)(ws + OFF_H1);
    float*          mpart  = (float*)(ws + OFF_MPART);
    float*          lpart  = (float*)(ws + OFF_LPART);
    float*          epart  = (float*)(ws + OFF_EPART);

    k_qproj <<<32, 256, 0, stream>>>(lat, wq, bq, qs);
    k_qwpack<<<64, 256, 0, stream>>>(qs, wk, bk, qwpack, qbp);
    k_main  <<<nblk, 512, 0, stream>>>(hseq, qwpack, qbp, mpart, lpart, epart, chunks);
    k_merge <<<64, 256, 0, stream>>>(mpart, lpart, epart, eg, lg, nblk);
    k_attnout<<<32, 256, 0, stream>>>(eg, lg, wv, bv, outpre);
    k_oproj <<<32, 256, 0, stream>>>(outpre, wo, bo, dout);
    k_ln    <<<8, 256, 0, stream>>>(dout, lng, lnb, hn);
    k_mlp1  <<<64, 256, 0, stream>>>(hn, w1, b1, h1);
    k_mlp2  <<<32, 256, 0, stream>>>(h1, w2, b2, dout);
    k_mvec  <<<4, 256, 0, stream>>>(dout, dout + MTOK * DMODEL);
}
